// HeteroGNNEncoder_22187801051829
// MI455X (gfx1250) — matrix operand order repaired and checked by reading, not compiled, not run on hardware
//
#include <hip/hip_runtime.h>
#include <hip/hip_bf16.h>

// CDNA5 / gfx1250: wave32; v_wmma_f32_16x16x32_bf16 for all dense matmuls.
// A-tiles staged in LDS (shared by the 4 waves of a block), C-tiles staged in
// LDS for coalesced float4 global stores. Full-tile fast paths avoid per-element
// exec-mask predication (all N here are multiples of 16).

typedef __attribute__((ext_vector_type(16))) __bf16 v16bf;
typedef __attribute__((ext_vector_type(8)))  float  v8f;

#define HDIM 64
#define PAD  68   // LDS row stride in floats: 272B, 16B-aligned, bank-spreading

__device__ __forceinline__ v8f wmma_bf16(v16bf a, v16bf b, v8f c) {
  return __builtin_amdgcn_wmma_f32_16x16x32_bf16(false, a, false, b, (short)0, c, false, false);
}

// Build A-fragment (16x32 bf16) from an LDS tile (16 rows x PAD floats).
// Fragment element e: row = lane%16; K = e + 8*(lane/16) + 32*kt  (e<8)
//                                    K = 16 + (e-8) + 8*(lane/16) + 32*kt (e>=8)
__device__ __forceinline__ v16bf lds_a_frag(const float* __restrict__ sm, int nl, int half, int kt) {
  const float* p = sm + nl * PAD + 8 * half + 32 * kt;
  float4 q0 = *(const float4*)(p);
  float4 q1 = *(const float4*)(p + 4);
  float4 q2 = *(const float4*)(p + 16);
  float4 q3 = *(const float4*)(p + 20);
  v16bf a;
  a[0]  = (__bf16)q0.x; a[1]  = (__bf16)q0.y; a[2]  = (__bf16)q0.z; a[3]  = (__bf16)q0.w;
  a[4]  = (__bf16)q1.x; a[5]  = (__bf16)q1.y; a[6]  = (__bf16)q1.z; a[7]  = (__bf16)q1.w;
  a[8]  = (__bf16)q2.x; a[9]  = (__bf16)q2.y; a[10] = (__bf16)q2.z; a[11] = (__bf16)q2.w;
  a[12] = (__bf16)q3.x; a[13] = (__bf16)q3.y; a[14] = (__bf16)q3.z; a[15] = (__bf16)q3.w;
  return a;
}

// ---------------------------------------------------------------------------
// Pack 28 f32 weight matrices ([K x 64], K<=64, zero-padded) into bf16
// B-fragment layout: frag[mat][kt(2)][ct(4)][lane(32)][i(16)]
//   lane%16 = N col in ct-tile, element i: K = kt*32 + 8*(lane/16) + (i%8) + 16*(i/8)
// ---------------------------------------------------------------------------
__global__ void pack_weights_kernel(const float* __restrict__ wp_u, const float* __restrict__ wp_r,
                                    const float* __restrict__ wp_d, const float* __restrict__ wp_i,
                                    const float* __restrict__ lin_l, const float* __restrict__ lin_r,
                                    __bf16* __restrict__ frags) {
  int idx = blockIdx.x * blockDim.x + threadIdx.x;
  const int total = 28 * 4096;
  if (idx >= total) return;
  int i    = idx & 15;
  int lane = (idx >> 4) & 31;
  int ct   = (idx >> 9) & 3;
  int kt   = (idx >> 11) & 1;
  int mat  = idx >> 12;
  int k = kt * 32 + (lane >> 4) * 8 + (i & 7) + (i >> 3) * 16;
  int n = ct * 16 + (lane & 15);
  const float* w;
  int kin;
  if (mat == 0)      { w = wp_u; kin = 32; }
  else if (mat == 1) { w = wp_r; kin = 16; }
  else if (mat == 2) { w = wp_d; kin = 8;  }
  else if (mat == 3) { w = wp_i; kin = 8;  }
  else if (mat < 16) { w = lin_l + (size_t)(mat - 4)  * 4096; kin = 64; }
  else               { w = lin_r + (size_t)(mat - 16) * 4096; kin = 64; }
  float f = (k < kin) ? w[(size_t)k * HDIM + n] : 0.0f;
  frags[idx] = (__bf16)f;
}

__global__ void zero_kernel(float* __restrict__ p, long long n) {
  long long i = (long long)blockIdx.x * blockDim.x + threadIdx.x;
  long long step = (long long)gridDim.x * blockDim.x;
  for (; i < n; i += step) p[i] = 0.0f;
}

// 32 lanes per edge: float2 gather (coalesced 256B/edge) + f32 atomic scatter.
__global__ void scatter_kernel(const int* __restrict__ src, const int* __restrict__ dst,
                               const float* __restrict__ hsrc, float* __restrict__ agg,
                               float* __restrict__ deg, int E) {
  long long tid = (long long)blockIdx.x * blockDim.x + threadIdx.x;
  int e = (int)(tid >> 5);
  int lane = (int)(tid & 31);
  if (e >= E) return;
  int s = src[e];
  int d = dst[e];
  const float2 v = *(const float2*)(hsrc + (size_t)s * HDIM + lane * 2);
  atomicAdd(agg + (size_t)d * HDIM + lane * 2,     v.x);
  atomicAdd(agg + (size_t)d * HDIM + lane * 2 + 1, v.y);
  if (lane == 0) atomicAdd(deg + d, 1.0f);
}

// h = x[N x kin] @ Wp + bp. Block = 128 threads (4 waves, one per 16-col tile).
__global__ void proj_kernel(const float* __restrict__ x, int kin, const __bf16* __restrict__ wfrag,
                            const float* __restrict__ bias, float* __restrict__ hout, int N) {
  __shared__ float smX[16 * PAD];
  int t    = threadIdx.x;
  int lane = t & 31;
  int ct   = t >> 5;
  int row0 = blockIdx.x * 16;
  bool full = (row0 + 16) <= N;
  int half = lane >> 4, nl = lane & 15;

  // cooperative load of x tile (16 x kin floats) into LDS
  int nq = kin >> 2;             // float4 per row (2/4/8)
  if (t < 16 * nq) {
    int row = t / nq;
    int c4  = (t - row * nq) * 4;
    float4 v = make_float4(0.f, 0.f, 0.f, 0.f);
    if (full || row0 + row < N) v = *(const float4*)(x + (size_t)(row0 + row) * kin + c4);
    *(float4*)(smX + row * PAD + c4) = v;
  }
  __syncthreads();

  float bv = bias[ct * 16 + nl];
  v8f c;
#pragma unroll
  for (int v = 0; v < 8; ++v) c[v] = bv;

  // A fragment (K<=32, single step); zero beyond kin
  v16bf a;
#pragma unroll
  for (int e = 0; e < 16; ++e) {
    int k = ((e < 8) ? e : (8 + e)) + 8 * half;   // e<8 -> e ; e>=8 -> 16+(e-8)
    float f = (k < kin) ? smX[nl * PAD + k] : 0.0f;
    a[e] = (__bf16)f;
  }
  v16bf b = ((const v16bf*)(wfrag + (size_t)ct * 512))[lane];
  c = wmma_bf16(a, b, c);

  __syncthreads();
#pragma unroll
  for (int v = 0; v < 8; ++v) smX[(v + 8 * half) * PAD + ct * 16 + nl] = c[v];
  __syncthreads();

#pragma unroll
  for (int chunk = 0; chunk < 2; ++chunk) {
    int idx = chunk * 128 + t;                     // 0..255 float4 of the 16x64 tile
    int row = idx >> 4;
    int c4  = (idx & 15) << 2;
    if (full || row0 + row < N)
      *(float4*)(hout + (size_t)(row0 + row) * HDIM + c4) = *(const float4*)(smX + row * PAD + c4);
  }
}

// m_out (+)= (agg/max(deg,1)) @ Wl + b + h_dst @ Wr for one relation.
__global__ void sage_kernel(const float* __restrict__ agg, const float* __restrict__ deg,
                            const float* __restrict__ hdst, const __bf16* __restrict__ wl,
                            const __bf16* __restrict__ wr, const float* __restrict__ bias,
                            float* __restrict__ mout, int N, int accumulate) {
  __shared__ float smM[16 * PAD];   // mean tile; reused for C staging
  __shared__ float smH[16 * PAD];   // h_dst tile
  int t    = threadIdx.x;
  int lane = t & 31;
  int ct   = t >> 5;
  int row0 = blockIdx.x * 16;
  bool full = (row0 + 16) <= N;
  int half = lane >> 4, nl = lane & 15;

  float bv = bias[ct * 16 + nl];
  v8f c;
  if (accumulate) {
    // stage previous relation sum for C init
#pragma unroll
    for (int chunk = 0; chunk < 2; ++chunk) {
      int idx = chunk * 128 + t;
      int row = idx >> 4;
      int c4  = (idx & 15) << 2;
      float4 v = make_float4(0.f, 0.f, 0.f, 0.f);
      if (full || row0 + row < N) v = *(const float4*)(mout + (size_t)(row0 + row) * HDIM + c4);
      *(float4*)(smM + row * PAD + c4) = v;
    }
    __syncthreads();
#pragma unroll
    for (int v = 0; v < 8; ++v) c[v] = bv + smM[(v + 8 * half) * PAD + ct * 16 + nl];
    __syncthreads();
  } else {
#pragma unroll
    for (int v = 0; v < 8; ++v) c[v] = bv;
  }

  // cooperative stage of mean (deg-scaled) and h_dst tiles; shared by all 4 waves
#pragma unroll
  for (int chunk = 0; chunk < 2; ++chunk) {
    int idx = chunk * 128 + t;
    int row = idx >> 4;
    int c4  = (idx & 15) << 2;
    int gr  = row0 + row;
    float4 a = make_float4(0.f, 0.f, 0.f, 0.f);
    float4 h = a;
    if (full || gr < N) {
      float ivd = 1.0f / fmaxf(deg[gr], 1.0f);
      float4 aa = *(const float4*)(agg + (size_t)gr * HDIM + c4);
      a.x = aa.x * ivd; a.y = aa.y * ivd; a.z = aa.z * ivd; a.w = aa.w * ivd;
      h = *(const float4*)(hdst + (size_t)gr * HDIM + c4);
    }
    *(float4*)(smM + row * PAD + c4) = a;
    *(float4*)(smH + row * PAD + c4) = h;
  }
  __syncthreads();

#pragma unroll
  for (int kt = 0; kt < 2; ++kt) {
    v16bf am = lds_a_frag(smM, nl, half, kt);
    v16bf bl = ((const v16bf*)(wl + (size_t)(kt * 4 + ct) * 512))[lane];
    c = wmma_bf16(am, bl, c);
    v16bf ah = lds_a_frag(smH, nl, half, kt);
    v16bf br = ((const v16bf*)(wr + (size_t)(kt * 4 + ct) * 512))[lane];
    c = wmma_bf16(ah, br, c);
  }

  __syncthreads();
#pragma unroll
  for (int v = 0; v < 8; ++v) smM[(v + 8 * half) * PAD + ct * 16 + nl] = c[v];
  __syncthreads();

#pragma unroll
  for (int chunk = 0; chunk < 2; ++chunk) {
    int idx = chunk * 128 + t;
    int row = idx >> 4;
    int c4  = (idx & 15) << 2;
    if (full || row0 + row < N)
      *(float4*)(mout + (size_t)(row0 + row) * HDIM + c4) = *(const float4*)(smM + row * PAD + c4);
  }
}

// Channel-wise sum / sum-of-squares of m*inv_n.
__global__ void bn_stats_kernel(const float* __restrict__ m, int N, float inv_n,
                                float* __restrict__ stats) {
  int c = threadIdx.x & 63;
  int rowStart = (int)(((long long)blockIdx.x * blockDim.x + threadIdx.x) >> 6);
  int rowStep  = (gridDim.x * blockDim.x) >> 6;
  float s = 0.0f, s2 = 0.0f;
  for (int r = rowStart; r < N; r += rowStep) {
    float v = m[(size_t)r * HDIM + c] * inv_n;
    s += v;
    s2 += v * v;
  }
  atomicAdd(stats + c, s);
  atomicAdd(stats + 64 + c, s2);
}

// y = relu( (m*inv_n - mu)/sqrt(var+eps) * gamma + beta ), in place.
__global__ void bn_apply_kernel(float* __restrict__ m, const float* __restrict__ stats,
                                const float* __restrict__ gamma, const float* __restrict__ beta,
                                int N, float inv_n) {
  long long tid = (long long)blockIdx.x * blockDim.x + threadIdx.x;
  if (tid >= (long long)N * HDIM) return;
  int c = (int)(tid & 63);
  float mu  = stats[c] / (float)N;
  float var = stats[64 + c] / (float)N - mu * mu;
  float v = m[tid] * inv_n;
  float y = (v - mu) * rsqrtf(var + 1e-5f) * gamma[c] + beta[c];
  m[tid] = fmaxf(y, 0.0f);
}

// ---------------------------------------------------------------------------

extern "C" void kernel_launch(void* const* d_in, const int* in_sizes, int n_in,
                              void* d_out, int out_size, void* d_ws, size_t ws_size,
                              hipStream_t stream) {
  const float* x[4] = {(const float*)d_in[0], (const float*)d_in[1],
                       (const float*)d_in[2], (const float*)d_in[3]};
  const int xdim[4] = {32, 16, 8, 8};
  int Ns[4];
  for (int t = 0; t < 4; ++t) Ns[t] = in_sizes[t] / xdim[t];

  const float* Wp[4] = {(const float*)d_in[16], (const float*)d_in[18],
                        (const float*)d_in[20], (const float*)d_in[22]};
  const float* bp[4] = {(const float*)d_in[17], (const float*)d_in[19],
                        (const float*)d_in[21], (const float*)d_in[23]};
  const float* lin_l_W  = (const float*)d_in[24];
  const float* lin_l_b  = (const float*)d_in[25];
  const float* lin_r_W  = (const float*)d_in[26];
  const float* bn_gamma = (const float*)d_in[27];
  const float* bn_beta  = (const float*)d_in[28];

  size_t hoff[4];
  size_t szH = 0;
  for (int t = 0; t < 4; ++t) { hoff[t] = szH; szH += (size_t)Ns[t] * HDIM; }
  int maxN = Ns[0];
  for (int t = 1; t < 4; ++t) if (Ns[t] > maxN) maxN = Ns[t];

  float* ws = (float*)d_ws;
  size_t off = 0;
  float* hA    = ws + off; off += szH;
  float* hB    = ws + off; off += szH;
  float* agg   = ws + off; off += (size_t)maxN * HDIM;
  float* deg   = ws + off; off += ((size_t)maxN + 63) & ~(size_t)63;
  float* stats = ws + off; off += 128;
  __bf16* frags = (__bf16*)(ws + off);
  size_t need = off * sizeof(float) + (size_t)28 * 4096 * sizeof(__bf16);
  if (ws_size < need) return;  // deterministic no-op if scratch undersized

  {
    int total = 28 * 4096;
    pack_weights_kernel<<<(total + 255) / 256, 256, 0, stream>>>(
        Wp[0], Wp[1], Wp[2], Wp[3], lin_l_W, lin_r_W, frags);
  }

  for (int t = 0; t < 4; ++t) {
    proj_kernel<<<(Ns[t] + 15) / 16, 128, 0, stream>>>(
        x[t], xdim[t], frags + (size_t)t * 4096, bp[t], hA + hoff[t], Ns[t]);
  }

  struct RelInfo { int si, di, st, dt; };
  const RelInfo R[6] = {{4, 5, 0, 1},  {6, 7, 1, 0},  {8, 9, 0, 2},
                        {10, 11, 2, 0}, {12, 13, 0, 3}, {14, 15, 3, 0}};

  float* hcur = hA;
  for (int l = 0; l < 2; ++l) {
    float* hnext = (l == 0) ? hB : (float*)d_out;
    for (int t = 0; t < 4; ++t) {
      int Nt = Ns[t];
      int cnt = 0;
      for (int r = 0; r < 6; ++r) {
        if (R[r].dt != t) continue;
        int E = in_sizes[R[r].si];
        long long aggN = (long long)Nt * HDIM;
        int zb = (int)((aggN + 255) / 256); if (zb > 2048) zb = 2048;
        zero_kernel<<<zb, 256, 0, stream>>>(agg, aggN);
        zero_kernel<<<(Nt + 255) / 256, 256, 0, stream>>>(deg, Nt);
        long long sthreads = (long long)E * 32;
        scatter_kernel<<<(int)((sthreads + 255) / 256), 256, 0, stream>>>(
            (const int*)d_in[R[r].si], (const int*)d_in[R[r].di],
            hcur + hoff[R[r].st], agg, deg, E);
        sage_kernel<<<(Nt + 15) / 16, 128, 0, stream>>>(
            agg, deg, hcur + hoff[t],
            frags + (size_t)(4 + l * 6 + r) * 4096,
            frags + (size_t)(16 + l * 6 + r) * 4096,
            lin_l_b + (size_t)(l * 6 + r) * HDIM,
            hnext + hoff[t], Nt, cnt);
        ++cnt;
      }
      float inv_n = 1.0f / (float)cnt;
      zero_kernel<<<1, 128, 0, stream>>>(stats, 128);
      bn_stats_kernel<<<256, 256, 0, stream>>>(hnext + hoff[t], Nt, inv_n, stats);
      long long etotal = (long long)Nt * HDIM;
      bn_apply_kernel<<<(int)((etotal + 255) / 256), 256, 0, stream>>>(
          hnext + hoff[t], stats,
          bn_gamma + (size_t)(l * 4 + t) * HDIM,
          bn_beta + (size_t)(l * 4 + t) * HDIM, Nt, inv_n);
    }
    hcur = hnext;
  }
}
